// Agent_24550033063971
// MI455X (gfx1250) — compile-verified, matrix operand order checked by reading, not run on hardware
//
#include <hip/hip_runtime.h>
#include <hip/hip_bf16.h>
#include <math.h>

// ---------------------------------------------------------------------------
// CDNA5 (gfx1250) conv+LSTM agent forward pass.
// All GEMMs run on V_WMMA_F32_16X16X32_BF16 (fp32 accumulate). Weights are
// converted to bf16 once per call; input-projection weights are concatenated
// onto the recurrent weights along K so each LSTM step is ONE WMMA GEMM:
//      gates[t] = [h_masked | x_t] @ [Whh | Wih]^T + (bih + bhh)
// GEMM: MT(2, M) x NT(4, N) tile register blocking per wave -> 8 independent
// WMMA chains (no RAW hazard NOPs), 0.75 fragment loads per WMMA, and the
// dominant L2 weight stream is re-read only M/32 times per step instead of
// M/16 (the recurrent step is L2-BW bound, not math bound). K1/K2 are
// template constants so the K loop fully unrolls with static VMEM offsets.
// ---------------------------------------------------------------------------

typedef __bf16 bf16_t;
typedef __attribute__((ext_vector_type(16))) __bf16 v16bf;
typedef __attribute__((ext_vector_type(8)))  __bf16 v8bf;
typedef __attribute__((ext_vector_type(8)))  float  v8f;

#define T_STEPS 64
#define BATCH   128
#define HID     1024
#define ENC_H   256

// ---------------------------------------------------------------------------
// Fragment loader: 16-bit A/B operand, 16x32 tile, wave32 layout (ISA 7.12.2):
//   lanes 0-15 : row = lane,    elems 0-7 = K 0-7,  elems 8-15 = K 16-23
//   lanes 16-31: row = lane-16, elems 0-7 = K 8-15, elems 8-15 = K 24-31
// rowp points at this lane's row; khalf = (lane>=16) ? 8 : 0.
// All row strides are multiples of 32 elems (64B) -> 16B-aligned v8bf loads.
// ---------------------------------------------------------------------------
__device__ __forceinline__ v16bf load_frag16(const bf16_t* __restrict__ rowp,
                                             int k0, int khalf) {
  v8bf lo = *(const v8bf*)(rowp + k0 + khalf);
  v8bf hi = *(const v8bf*)(rowp + k0 + 16 + khalf);
  v16bf r;
#pragma unroll
  for (int e = 0; e < 8; ++e) { r[e] = lo[e]; r[e + 8] = hi[e]; }
  return r;
}

// ---------------------------------------------------------------------------
// WMMA GEMM:  C[M,N] = act( [A1 | A2][M, K1+K2] @ W[N,K]^T + b0 + b1 )
// (16*MT) x (16*NT) output block per wave; 8 waves (256 threads) per block.
// W is [N, K] row-major bf16 (torch weight layout), so the B-operand gather
// is the same per-lane pattern as A. Output per ISA C/D layout.
// ---------------------------------------------------------------------------
template <int OUT_BF16, int RELU, int MT, int NT, int K1, int K2>
__global__ void gemm_wmma_bf16(const bf16_t* __restrict__ A1,
                               const bf16_t* __restrict__ A2,
                               const bf16_t* __restrict__ W,
                               const float* __restrict__ b0,
                               const float* __restrict__ b1,
                               void* __restrict__ Cout, int ldc, int coff,
                               int M, int N) {
  constexpr int K = K1 + K2;
  const int lane    = threadIdx.x & 31;
  const int wave    = (blockIdx.x * blockDim.x + threadIdx.x) >> 5;
  const int stripsN = N / (16 * NT);
  const int tb = wave / stripsN;   // M block index (MT tiles)
  const int ts = wave - tb * stripsN;
  if (tb >= M / (16 * MT)) return;  // whole-wave exit: EXEC all-ones for WMMA

  const int r     = lane & 15;
  const int khalf = (lane >> 4) << 3;  // 0 or 8
  const bf16_t* a1p[MT];
  const bf16_t* a2p[MT];
#pragma unroll
  for (int mi = 0; mi < MT; ++mi) {
    const int row = (tb * MT + mi) * 16 + r;
    a1p[mi] = A1 + (size_t)row * K1;
    a2p[mi] = (K2 > 0) ? (A2 + (size_t)row * K2) : A1;
  }
  const bf16_t* wp = W + (size_t)(ts * (16 * NT) + r) * K;  // first of NT rows

  v8f acc[MT][NT];
#pragma unroll
  for (int mi = 0; mi < MT; ++mi)
#pragma unroll
    for (int j = 0; j < NT; ++j) acc[mi][j] = (v8f){};

  // Fully unrolled K loop; straight-line code + 8 independent WMMA chains let
  // the scheduler hoist next-chunk loads over current-chunk WMMAs.
#pragma unroll
  for (int k0 = 0; k0 < K; k0 += 32) {
    __builtin_prefetch((const void*)(wp + k0 + 96), 0, 1);  // global_prefetch_b8
    v16bf af[MT];
#pragma unroll
    for (int mi = 0; mi < MT; ++mi)
      af[mi] = (k0 < K1) ? load_frag16(a1p[mi], k0, khalf)
                         : load_frag16(a2p[mi], k0 - K1, khalf);
    v16bf bfr[NT];
#pragma unroll
    for (int j = 0; j < NT; ++j)
      bfr[j] = load_frag16(wp + (size_t)(16 * j) * K, k0, khalf);
#pragma unroll
    for (int j = 0; j < NT; ++j)
#pragma unroll
      for (int mi = 0; mi < MT; ++mi)
        acc[mi][j] = __builtin_amdgcn_wmma_f32_16x16x32_bf16(
            false, af[mi], false, bfr[j], (short)0, acc[mi][j], false, false);
  }

#pragma unroll
  for (int mi = 0; mi < MT; ++mi) {
    const int mbase = (tb * MT + mi) * 16 + ((lane >> 4) << 3);
#pragma unroll
    for (int j = 0; j < NT; ++j) {
      const int n = ts * (16 * NT) + j * 16 + (lane & 15);
      float bias = 0.0f;
      if (b0) bias += b0[n];
      if (b1) bias += b1[n];
#pragma unroll
      for (int e = 0; e < 8; ++e) {
        float v = acc[mi][j][e] + bias;
        if (RELU) v = v > 0.0f ? v : 0.0f;
        size_t offo = (size_t)(mbase + e) * ldc + coff + n;
        if (OUT_BF16) ((bf16_t*)Cout)[offo] = (bf16_t)v;
        else          ((float*)Cout)[offo]  = v;
      }
    }
  }
}

// ---------------------------------------------------------------------------
// fp32 -> bf16 row-wise convert with destination stride/offset (weight packer)
// ---------------------------------------------------------------------------
__global__ void convert_rows_bf16(const float* __restrict__ src,
                                  bf16_t* __restrict__ dst,
                                  int rows, int cols, int dst_stride, int dst_off) {
  int i = blockIdx.x * blockDim.x + threadIdx.x;
  if (i >= rows * cols) return;
  int r = i / cols, c = i - r * cols;
  dst[(size_t)r * dst_stride + dst_off + c] = (bf16_t)src[i];
}

__global__ void head_bias_pack(const float* __restrict__ ab,
                               const float* __restrict__ cb,
                               float* __restrict__ out) {
  int j = threadIdx.x;
  if (j < 15) out[j] = ab[j];
  else if (j == 15) out[j] = cb[0];
}

// ---------------------------------------------------------------------------
// Fused conv1(9x9,s9) -> conv2(3x3) -> conv3(3x3), ReLU each, one block/image.
// Dynamic LDS: image(9801) + c1(1936) + c2(2592) floats = 57316 B.
// Flatten order of out = [oc*49 + oy*7 + ox] to match fc_w.
// ---------------------------------------------------------------------------
__global__ void conv_stack(const float* __restrict__ img,
                           const float* __restrict__ w1, const float* __restrict__ b1,
                           const float* __restrict__ w2, const float* __restrict__ b2,
                           const float* __restrict__ w3, const float* __restrict__ b3,
                           bf16_t* __restrict__ out) {
  extern __shared__ float smem[];
  float* simg = smem;           // 99*99
  float* c1   = smem + 9801;    // 16*11*11
  float* c2   = c1 + 1936;      // 32*9*9
  const int tid = threadIdx.x;
  const float* ip = img + (size_t)blockIdx.x * 9801;
  for (int j = tid; j < 9801; j += blockDim.x) simg[j] = ip[j] * (1.0f / 255.0f);
  __syncthreads();

  for (int o = tid; o < 16 * 121; o += blockDim.x) {
    int oc = o / 121, rem = o - oc * 121, oy = rem / 11, ox = rem - (rem / 11) * 11;
    float s = b1[oc];
    const float* wp   = w1 + oc * 81;
    const float* base = simg + (oy * 9) * 99 + ox * 9;
#pragma unroll
    for (int ky = 0; ky < 9; ++ky)
#pragma unroll
      for (int kx = 0; kx < 9; ++kx)
        s += base[ky * 99 + kx] * wp[ky * 9 + kx];
    c1[o] = s > 0.0f ? s : 0.0f;
  }
  __syncthreads();

  for (int o = tid; o < 32 * 81; o += blockDim.x) {
    int oc = o / 81, rem = o - oc * 81, oy = rem / 9, ox = rem - (rem / 9) * 9;
    float s = b2[oc];
    for (int ic = 0; ic < 16; ++ic) {
      const float* base = c1 + ic * 121 + oy * 11 + ox;
      const float* wp   = w2 + (oc * 16 + ic) * 9;
#pragma unroll
      for (int ky = 0; ky < 3; ++ky)
#pragma unroll
        for (int kx = 0; kx < 3; ++kx)
          s += base[ky * 11 + kx] * wp[ky * 3 + kx];
    }
    c2[o] = s > 0.0f ? s : 0.0f;
  }
  __syncthreads();

  bf16_t* op = out + (size_t)blockIdx.x * 1568;
  for (int o = tid; o < 32 * 49; o += blockDim.x) {
    int oc = o / 49, rem = o - oc * 49, oy = rem / 7, ox = rem - (rem / 7) * 7;
    float s = b3[oc];
    for (int ic = 0; ic < 32; ++ic) {
      const float* base = c2 + ic * 81 + oy * 9 + ox;
      const float* wp   = w3 + (oc * 32 + ic) * 9;
#pragma unroll
      for (int ky = 0; ky < 3; ++ky)
#pragma unroll
        for (int kx = 0; kx < 3; ++kx)
          s += base[ky * 9 + kx] * wp[ky * 3 + kx];
    }
    op[o] = (bf16_t)(s > 0.0f ? s : 0.0f);
  }
}

// ---------------------------------------------------------------------------
// Language embedding gather -> bf16 [T*B, 32]
// ---------------------------------------------------------------------------
__global__ void embed_lang(const int* __restrict__ ids,
                           const float* __restrict__ emb,
                           bf16_t* __restrict__ out, int n) {
  int i = blockIdx.x * blockDim.x + threadIdx.x;
  if (i >= n) return;
  int row = i >> 5, col = i & 31;
  out[i] = (bf16_t)(emb[ids[row] * 32 + col]);
}

// ---------------------------------------------------------------------------
// LSTM step prologue: h,c masked by (1 - done); h emitted as bf16 for WMMA.
// per_env=0 -> scalar mask done[mask_base]; per_env=1 -> done[mask_base + b].
// ---------------------------------------------------------------------------
__global__ void lstm_prep(const float* __restrict__ done, int mask_base,
                          int per_env, int hid,
                          const float* __restrict__ h,
                          const float* __restrict__ cin,
                          float* __restrict__ cout,
                          bf16_t* __restrict__ hbf, int n) {
  int i = blockIdx.x * blockDim.x + threadIdx.x;
  if (i >= n) return;
  float m = 1.0f - done[mask_base + (per_env ? (i / hid) : 0)];
  hbf[i]  = (bf16_t)(h[i] * m);
  cout[i] = cin[i] * m;
}

__device__ __forceinline__ float sigm(float x) { return 1.0f / (1.0f + __expf(-x)); }

// ---------------------------------------------------------------------------
// LSTM pointwise cell: gates [B, 4*hid] order (i,f,g,o); updates h,c fp32 and
// writes h as bf16 into the per-step sequence buffer (feeds the next GEMM).
// ---------------------------------------------------------------------------
__global__ void lstm_cell(const float* __restrict__ gates, int hid,
                          float* __restrict__ c, float* __restrict__ h,
                          bf16_t* __restrict__ hs, int n) {
  int i = blockIdx.x * blockDim.x + threadIdx.x;
  if (i >= n) return;
  int b = i / hid, j = i - b * hid;
  const float* g = gates + (size_t)b * 4 * hid;
  float gi = g[j], gf = g[hid + j], gg = g[2 * hid + j], go = g[3 * hid + j];
  float cn = sigm(gf) * c[i] + sigm(gi) * tanhf(gg);
  float hn = sigm(go) * tanhf(cn);
  c[i] = cn;
  h[i] = hn;
  hs[i] = (bf16_t)hn;
}

// ---------------------------------------------------------------------------
extern "C" void kernel_launch(void* const* d_in, const int* in_sizes, int n_in,
                              void* d_out, int out_size, void* d_ws, size_t ws_size,
                              hipStream_t stream) {
  (void)in_sizes; (void)n_in; (void)out_size; (void)ws_size;
  const float* x_img   = (const float*)d_in[0];
  const int*   x_lang  = (const int*)d_in[1];
  const float* done    = (const float*)d_in[2];
  const float* enc_h0  = (const float*)d_in[3];
  const float* enc_c0  = (const float*)d_in[4];
  const float* mem_h0  = (const float*)d_in[5];
  const float* mem_c0  = (const float*)d_in[6];
  const float* emb     = (const float*)d_in[7];
  const float* conv1_w = (const float*)d_in[8];
  const float* conv1_b = (const float*)d_in[9];
  const float* conv2_w = (const float*)d_in[10];
  const float* conv2_b = (const float*)d_in[11];
  const float* conv3_w = (const float*)d_in[12];
  const float* conv3_b = (const float*)d_in[13];
  const float* fc_w    = (const float*)d_in[14];
  const float* fc_b    = (const float*)d_in[15];
  const float* enc_Wih = (const float*)d_in[16];
  const float* enc_Whh = (const float*)d_in[17];
  const float* enc_bih = (const float*)d_in[18];
  const float* enc_bhh = (const float*)d_in[19];
  const float* lemb_w  = (const float*)d_in[20];
  const float* lemb_b  = (const float*)d_in[21];
  const float* mem_Wih = (const float*)d_in[22];
  const float* mem_Whh = (const float*)d_in[23];
  const float* mem_bih = (const float*)d_in[24];
  const float* mem_bhh = (const float*)d_in[25];
  const float* actor_w = (const float*)d_in[26];
  const float* actor_b = (const float*)d_in[27];
  const float* critic_w = (const float*)d_in[28];
  const float* critic_b = (const float*)d_in[29];

  const int TB = T_STEPS * BATCH;  // 8192

  // ---- workspace carve-up (256B aligned) ----
  char* base = (char*)d_ws;
  size_t off = 0;
  auto carve = [&](size_t bytes) {
    char* p = base + off;
    off += (bytes + 255) & ~(size_t)255;
    return p;
  };
  bf16_t* convout  = (bf16_t*)carve((size_t)TB * 1568 * 2);   // conv features
  bf16_t* hidden   = (bf16_t*)carve((size_t)TB * 288 * 2);    // [img(256)|lang(32)]
  bf16_t* lang_in  = (bf16_t*)carve((size_t)TB * 32 * 2);
  bf16_t* hs_enc   = (bf16_t*)carve((size_t)TB * ENC_H * 2);
  bf16_t* hs_mem   = (bf16_t*)carve((size_t)TB * HID * 2);
  bf16_t* fcw      = (bf16_t*)carve((size_t)256 * 1568 * 2);
  bf16_t* Wenc     = (bf16_t*)carve((size_t)4 * ENC_H * 288 * 2);    // [1024, 256+32]
  bf16_t* lembw    = (bf16_t*)carve((size_t)32 * 256 * 2);
  bf16_t* Wmem     = (bf16_t*)carve((size_t)4 * HID * 1312 * 2);     // [4096, 1024+288]
  bf16_t* headw    = (bf16_t*)carve((size_t)16 * HID * 2);
  float*  headb    = (float*) carve(16 * 4);
  float*  gates_e  = (float*) carve((size_t)BATCH * 4 * ENC_H * 4);
  float*  gates_m  = (float*) carve((size_t)BATCH * 4 * HID * 4);
  float*  h_enc    = (float*) carve((size_t)BATCH * ENC_H * 4);
  float*  c_enc    = (float*) carve((size_t)BATCH * ENC_H * 4);
  float*  h_mem    = (float*) carve((size_t)BATCH * HID * 4);
  float*  c_mem    = (float*) carve((size_t)BATCH * HID * 4);
  bf16_t* hbf_enc  = (bf16_t*)carve((size_t)BATCH * ENC_H * 2);
  bf16_t* hbf_mem  = (bf16_t*)carve((size_t)BATCH * HID * 2);

  auto cdiv = [](int a, int b) { return (a + b - 1) / b; };
  auto blocks = [&](int M, int N, int MT, int NT) {
    return cdiv((M / (16 * MT)) * (N / (16 * NT)), 8);
  };

  // ---- pack weights to bf16 (Wih concatenated after Whh along K) ----
  convert_rows_bf16<<<cdiv(256 * 1568, 256), 256, 0, stream>>>(fc_w, fcw, 256, 1568, 1568, 0);
  convert_rows_bf16<<<cdiv(1024 * 256, 256), 256, 0, stream>>>(enc_Whh, Wenc, 1024, 256, 288, 0);
  convert_rows_bf16<<<cdiv(1024 * 32, 256), 256, 0, stream>>>(enc_Wih, Wenc, 1024, 32, 288, 256);
  convert_rows_bf16<<<cdiv(32 * 256, 256), 256, 0, stream>>>(lemb_w, lembw, 32, 256, 256, 0);
  convert_rows_bf16<<<cdiv(4096 * 1024, 256), 256, 0, stream>>>(mem_Whh, Wmem, 4096, 1024, 1312, 0);
  convert_rows_bf16<<<cdiv(4096 * 288, 256), 256, 0, stream>>>(mem_Wih, Wmem, 4096, 288, 1312, 1024);
  convert_rows_bf16<<<cdiv(15 * 1024, 256), 256, 0, stream>>>(actor_w, headw, 15, 1024, 1024, 0);
  convert_rows_bf16<<<cdiv(1024, 256), 256, 0, stream>>>(critic_w, headw + (size_t)15 * 1024, 1, 1024, 1024, 0);
  head_bias_pack<<<1, 16, 0, stream>>>(actor_b, critic_b, headb);

  // ---- conv stack (one block per image), bf16 features ----
  conv_stack<<<TB, 256, (9801 + 1936 + 2592) * sizeof(float), stream>>>(
      x_img, conv1_w, conv1_b, conv2_w, conv2_b, conv3_w, conv3_b, convout);

  // ---- FC: img_hidden = relu(conv @ fc_w^T + fc_b) -> hidden cols [0,256) ----
  gemm_wmma_bf16<1, 1, 2, 4, 1568, 0><<<blocks(TB, 256, 2, 4), 256, 0, stream>>>(
      convout, nullptr, fcw, fc_b, nullptr, hidden, 288, 0, TB, 256);

  // ---- language embedding ----
  embed_lang<<<cdiv(TB * 32, 256), 256, 0, stream>>>(x_lang, emb, lang_in, TB * 32);

  // ---- encoder LSTM: scalar per-step mask 1-done[t] (reference quirk) ----
  {
    const int n = BATCH * ENC_H;                // 32768
    for (int t = 0; t < T_STEPS; ++t) {
      lstm_prep<<<cdiv(n, 256), 256, 0, stream>>>(
          done, t, 0, ENC_H,
          t == 0 ? enc_h0 : h_enc, t == 0 ? enc_c0 : c_enc,
          c_enc, hbf_enc, n);
      gemm_wmma_bf16<0, 0, 2, 4, ENC_H, 32>
          <<<blocks(BATCH, 4 * ENC_H, 2, 4), 256, 0, stream>>>(
          hbf_enc, lang_in + (size_t)t * BATCH * 32,
          Wenc, enc_bih, enc_bhh, gates_e, 4 * ENC_H, 0, BATCH, 4 * ENC_H);
      lstm_cell<<<cdiv(n, 256), 256, 0, stream>>>(
          gates_e, ENC_H, c_enc, h_enc, hs_enc + (size_t)t * n, n);
    }
  }

  // ---- lemb: lang_hidden = relu(hs_enc @ lemb_w^T + b) -> hidden cols [256,288) ----
  gemm_wmma_bf16<1, 1, 2, 2, 256, 0><<<blocks(TB, 32, 2, 2), 256, 0, stream>>>(
      hs_enc, nullptr, lembw, lemb_b, nullptr, hidden, 288, 256, TB, 32);

  // ---- memory LSTM: per-env mask 1-done[t*B+b]; K-fused [h|x_t] GEMM ----
  {
    const int n = BATCH * HID;                  // 131072
    for (int t = 0; t < T_STEPS; ++t) {
      lstm_prep<<<cdiv(n, 256), 256, 0, stream>>>(
          done, t * BATCH, 1, HID,
          t == 0 ? mem_h0 : h_mem, t == 0 ? mem_c0 : c_mem,
          c_mem, hbf_mem, n);
      gemm_wmma_bf16<0, 0, 2, 4, HID, 288>
          <<<blocks(BATCH, 4 * HID, 2, 4), 256, 0, stream>>>(
          hbf_mem, hidden + (size_t)t * BATCH * 288,
          Wmem, mem_bih, mem_bhh, gates_m, 4 * HID, 0, BATCH, 4 * HID);
      lstm_cell<<<cdiv(n, 256), 256, 0, stream>>>(
          gates_m, HID, c_mem, h_mem, hs_mem + (size_t)t * n, n);
    }
  }

  // ---- heads: [actor(15); critic(1)] = one 16-wide N tile -> d_out [TB,16] ----
  gemm_wmma_bf16<0, 0, 2, 1, HID, 0><<<blocks(TB, 16, 2, 1), 256, 0, stream>>>(
      hs_mem, nullptr, headw, headb, nullptr, (float*)d_out, 16, 0, TB, 16);
}
